// attention_mech_28664611733586
// MI455X (gfx1250) — matrix operand order repaired and checked by reading, not compiled, not run on hardware
//
#include <hip/hip_runtime.h>

// ---------------------------------------------------------------------------
// MI455X (gfx1250, wave32) attention-variant kernel.
// Heavy GEMMs: v_wmma_f32_16x16x32_bf16.  keys staging in stats kernel uses
// the Tensor Data Mover (tensor_load_to_lds + s_wait_tensorcnt) double-buffered.
// Pipeline: proj(keys) -> proj(queries) -> proj(qprojT) -> stats(M,1/D) -> out
// Workspace (~12.4 MB, L2-resident):
//   [0,4MB)   keys    bf16 [BH*2048][64]
//   [4,8MB)   queries bf16 [BH*2048][64]
//   [8,12MB)  qprojT  bf16 [BH][64][2048]   (t-major for contiguous B frags)
//   [12MB..)  M  f32 [BH*2048], Dinv f32 [BH*2048]
// ---------------------------------------------------------------------------

#define DI __device__ __forceinline__

typedef __bf16 bf16_t;
typedef bf16_t v16bf __attribute__((ext_vector_type(16)));
typedef bf16_t v8bf  __attribute__((ext_vector_type(8)));
typedef float  v8f   __attribute__((ext_vector_type(8)));
typedef unsigned int u32x4 __attribute__((ext_vector_type(4)));
typedef int          i32x4 __attribute__((ext_vector_type(4)));
typedef int          i32x8 __attribute__((ext_vector_type(8)));

constexpr int NK = 2048, NQ = 2048, SD = 64;
constexpr int BH = 2 * 8;
constexpr int KT = 64;              // staged keys rows per TDM tile

DI v8bf ld8bf(const bf16_t* p) { return *(const v8bf*)p; }
DI v8f  ld8f (const float*  p) { return *(const v8f*)p; }

DI v16bf cat16(v8bf a, v8bf b) {
  return __builtin_shufflevector(a, b, 0,1,2,3,4,5,6,7,8,9,10,11,12,13,14,15);
}

DI v16bf cvt16(v8f a, v8f b) {
  v16bf r;
#pragma unroll
  for (int i = 0; i < 8; ++i) { r[i] = (bf16_t)a[i]; r[i + 8] = (bf16_t)b[i]; }
  return r;
}

DI v8f zero8() { v8f r = {0.f,0.f,0.f,0.f,0.f,0.f,0.f,0.f}; return r; }

DI v8f wmma_bf(v16bf a, v16bf b, v8f c) {
  // D = A(16x32 bf16) x B(32x16 bf16) + C(16x16 f32)
  return __builtin_amdgcn_wmma_f32_16x16x32_bf16(false, a, false, b,
                                                 (short)0, c, false, false);
}

// A fragment (16x32 bf16): lane row m = lane&15, K = h*8..h*8+7, 16+h*8..
DI v16bf loadA(const bf16_t* row, int h) {
  return cat16(ld8bf(row + h * 8), ld8bf(row + 16 + h * 8));
}
// B fragment (32x16 bf16): lane col n, K = h*8..h*8+7, 16+h*8..
DI v16bf loadB(const bf16_t* row, int h) {
  return cat16(ld8bf(row + h * 8), ld8bf(row + 16 + h * 8));
}
DI v16bf loadB_f32(const float* row, int h) {
  return cvt16(ld8f(row + h * 8), ld8f(row + 16 + h * 8));
}

#if __has_builtin(__builtin_amdgcn_tensor_load_to_lds)
#define USE_TDM 1
#else
#define USE_TDM 0
#endif

#if USE_TDM
// TDM 2D tile load: `rows` rows of 32 dwords (64 bf16) from a row-major
// tensor with 32-dword stride, into contiguous LDS at lds_off.
// D# fields per CDNA5 ISA.
DI void tdm_load_tile(unsigned int lds_off, unsigned long long gaddr,
                      unsigned int rows) {
  const unsigned int td0 = 32;      // tensor/tile dim0 & stride, dword units
  u32x4 g0;
  g0[0] = 1u;                                            // count=1, user D#
  g0[1] = lds_off;                                       // lds_addr (bytes)
  g0[2] = (unsigned int)gaddr;                           // global_addr lo
  g0[3] = (unsigned int)((gaddr >> 32) & 0x1FFFFFFu) | (2u << 30);  // type=2
  i32x8 g1;
  g1[0] = (int)(2u << 16);                               // data_size=4B
  g1[1] = (int)((td0 & 0xFFFFu) << 16);                  // tensor_dim0 lo16
  g1[2] = (int)((td0 >> 16) | (((unsigned)NK & 0xFFFFu) << 16)); // tensor_dim1
  g1[3] = (int)(((unsigned)NK >> 16) | (td0 << 16));     // tile_dim0
  g1[4] = (int)(rows & 0xFFFFu);                         // tile_dim1 (dim2=0)
  g1[5] = (int)td0;                                      // tensor_dim0_stride
  g1[6] = 0;
  g1[7] = 0;
  i32x4 z4 = {0, 0, 0, 0};
#if defined(__clang_major__) && __clang_major__ >= 23
  i32x8 z8 = {0, 0, 0, 0, 0, 0, 0, 0};
  __builtin_amdgcn_tensor_load_to_lds(g0, g1, z4, z4, z8, 0);
#else
  __builtin_amdgcn_tensor_load_to_lds(g0, g1, z4, z4, 0);
#endif
}
#endif

// ---------------------------------------------------------------------------
// proj: Y(Rx64) = X(Rx64) @ W(64x64, row stride wstride)^T [+b]
// 8 waves / block, each wave owns a 16-row tile.
// ---------------------------------------------------------------------------
template <bool IN_F32, bool TRANS, bool BIAS>
__global__ __launch_bounds__(256) void proj_kernel(
    const float* __restrict__ Xf, const bf16_t* __restrict__ Xb,
    const float* __restrict__ W, int wstride, int woff,
    const float* __restrict__ bias, bf16_t* __restrict__ Y) {
  const int lane = threadIdx.x & 31;
  const int wave = threadIdx.x >> 5;
  const int n = lane & 15, h = lane >> 4;
  const int m0 = (blockIdx.x * 8 + wave) * 16;

  v16bf a0, a1;
  if (IN_F32) {
    const float* xr = Xf + (size_t)(m0 + n) * SD;
    a0 = cvt16(ld8f(xr + h * 8), ld8f(xr + 16 + h * 8));
    a1 = cvt16(ld8f(xr + 32 + h * 8), ld8f(xr + 48 + h * 8));
  } else {
    const bf16_t* xr = Xb + (size_t)(m0 + n) * SD;
    a0 = loadA(xr, h);
    a1 = loadA(xr + 32, h);
  }

#pragma unroll
  for (int nt = 0; nt < 4; ++nt) {
    const float* wr = W + (size_t)(nt * 16 + n) * wstride + woff;
    v16bf b0 = loadB_f32(wr, h);
    v16bf b1 = loadB_f32(wr + 32, h);
    v8f c = zero8();
    c = wmma_bf(a0, b0, c);
    c = wmma_bf(a1, b1, c);
    const float bb = BIAS ? bias[nt * 16 + n] : 0.0f;
    const int s = nt * 16 + n;
    if (TRANS) {
      v8bf pk;
#pragma unroll
      for (int j = 0; j < 8; ++j) pk[j] = (bf16_t)(c[j] + bb);
      const int m = m0 + h * 8;
      const int bh = m >> 11, ml = m & (NQ - 1);
      *(v8bf*)(Y + (size_t)bh * (SD * NQ) + (size_t)s * NQ + ml) = pk;
    } else {
#pragma unroll
      for (int j = 0; j < 8; ++j) {
        const int m = m0 + h * 8 + j;
        Y[(size_t)m * SD + s] = (bf16_t)(c[j] + bb);
      }
    }
  }
}

// ---------------------------------------------------------------------------
// stats: per q column, M[q] = max_k l[k,q], Dinv[q] = 1/sum_k exp(l-M).
// Keys tiles (64x64 bf16, 8KB) are staged into LDS by the Tensor Data Mover,
// double-buffered; all 8 waves consume them as WMMA A-fragments via ds loads.
// Column q lives in lanes (l, l^16) of the C fragment -> shfl_xor(16) combine.
// ---------------------------------------------------------------------------
__global__ __launch_bounds__(256) void stats_kernel(
    const bf16_t* __restrict__ keys, const bf16_t* __restrict__ queries,
    float* __restrict__ Mbuf, float* __restrict__ Dinv) {
  __shared__ __align__(16) bf16_t kbuf[2][KT * SD];   // 2 x 8KB
  const int lane = threadIdx.x & 31;
  const int wave = threadIdx.x >> 5;
  const int n = lane & 15, h = lane >> 4;
  const int bh = blockIdx.y;
  const int q0 = (blockIdx.x * 8 + wave) * 16;

  const bf16_t* qr = queries + ((size_t)bh * NQ + q0 + n) * SD;
  v16bf b0 = loadB(qr, h);
  v16bf b1 = loadB(qr + 32, h);

  const bf16_t* kbase = keys + (size_t)bh * NK * SD;
  constexpr int NT = NK / KT;       // 32 staged tiles

#if USE_TDM
  if (wave == 0) {
    tdm_load_tile((unsigned int)(size_t)&kbuf[0][0],
                  (unsigned long long)(size_t)kbase, KT);
  }
#endif

  float m = -INFINITY, d = 0.0f;
#pragma unroll 1
  for (int t = 0; t < NT; ++t) {
#if USE_TDM
    if (wave == 0) {
      if (t + 1 < NT) {
        tdm_load_tile((unsigned int)(size_t)&kbuf[(t + 1) & 1][0],
                      (unsigned long long)(size_t)(kbase + (size_t)(t + 1) * KT * SD),
                      KT);
        __builtin_amdgcn_s_wait_tensorcnt((short)1);  // oldest (tile t) done
      } else {
        __builtin_amdgcn_s_wait_tensorcnt((short)0);
      }
    }
#else
    {  // cooperative fallback staging of tile t into buffer t&1
      const v8bf* src = (const v8bf*)(kbase + (size_t)t * KT * SD);
      v8bf* dst = (v8bf*)&kbuf[t & 1][0];
      dst[threadIdx.x] = src[threadIdx.x];
      dst[threadIdx.x + 256] = src[threadIdx.x + 256];
    }
#endif
    __syncthreads();
    const bf16_t* kb = &kbuf[t & 1][0];
#pragma unroll
    for (int sub = 0; sub < 4; ++sub) {
      const bf16_t* krow = kb + (size_t)(sub * 16 + n) * SD;   // ds loads
      v16bf a0 = loadA(krow, h);
      v16bf a1 = loadA(krow + 32, h);
      v8f c = zero8();
      c = wmma_bf(a0, b0, c);
      c = wmma_bf(a1, b1, c);
      float tmax = c[0];
#pragma unroll
      for (int j = 1; j < 8; ++j) tmax = fmaxf(tmax, c[j]);
      tmax = fmaxf(tmax, __shfl_xor(tmax, 16, 32));
      const float mnew = fmaxf(m, tmax);
      float s = 0.0f;
#pragma unroll
      for (int j = 0; j < 8; ++j) s += __expf(c[j] - mnew);
      s += __shfl_xor(s, 16, 32);
      d = d * __expf(m - mnew) + s;
      m = mnew;
    }
    __syncthreads();
  }
  if (h == 0) {
    Mbuf[(size_t)bh * NQ + q0 + n] = m;
    Dinv[(size_t)bh * NQ + q0 + n] = 1.0f / d;
  }
}

// ---------------------------------------------------------------------------
// out: per 16-row k tile: loop q (32 wide):
//   P = exp(keys@queries^T - M[q]) * Dinv[q]  (bf16, via LDS C->A transpose)
//   acc += P @ qprojT ; rsum += P @ ones
// epilogue: kproj = keys_tile @ Wv_k^T (reuses resident A frags)
//   out = acc + rsum * (kproj + bv)
// The P LDS region is per-wave private: same-wave LDS ops are in-order
// (DScnt), so only a compiler scheduling fence is needed, no s_barrier.
// ---------------------------------------------------------------------------
__global__ __launch_bounds__(256) void out_kernel(
    const bf16_t* __restrict__ keys, const bf16_t* __restrict__ queries,
    const bf16_t* __restrict__ qprojT, const float* __restrict__ Mbuf,
    const float* __restrict__ Dinv, const float* __restrict__ Wv,
    const float* __restrict__ bv, float* __restrict__ out) {
  __shared__ __align__(16) bf16_t ldsP[8][16 * 32];  // per-wave 16x32 P tile
  const int lane = threadIdx.x & 31;
  const int wave = threadIdx.x >> 5;
  const int n = lane & 15, h = lane >> 4;
  const int bh = blockIdx.y;
  const int k0 = (blockIdx.x * 8 + wave) * 16;

  const bf16_t* kr = keys + ((size_t)bh * NK + k0 + n) * SD;
  v16bf a0 = loadA(kr, h);
  v16bf a1 = loadA(kr + 32, h);

  v16bf ones;
#pragma unroll
  for (int i = 0; i < 16; ++i) ones[i] = (bf16_t)1.0f;

  v8f acc0 = zero8(), acc1 = zero8(), acc2 = zero8(), acc3 = zero8();
  v8f rsum = zero8();

  const bf16_t* qbase = queries + (size_t)bh * NQ * SD;
  const bf16_t* vbase = qprojT + (size_t)bh * SD * NQ;
  const float* Mb = Mbuf + (size_t)bh * NQ;
  const float* Db = Dinv + (size_t)bh * NQ;
  bf16_t* myP = &ldsP[wave][0];

#pragma unroll 1
  for (int q = 0; q < NQ; q += 32) {
    {  // prefetch next q-step operands (global_prefetch_b8)
      const int qn = (q + 32) & (NQ - 1);
      __builtin_prefetch(qbase + (size_t)(qn + n) * SD, 0, 1);
      __builtin_prefetch(vbase + (size_t)n * NQ + qn, 0, 1);
    }
#pragma unroll
    for (int nt = 0; nt < 2; ++nt) {
      const bf16_t* qrow = qbase + (size_t)(q + nt * 16 + n) * SD;
      v16bf b0 = loadB(qrow, h);
      v16bf b1 = loadB(qrow + 32, h);
      v8f c = zero8();
      c = wmma_bf(a0, b0, c);
      c = wmma_bf(a1, b1, c);
      const float Mq = Mb[q + nt * 16 + n];
      const float Di = Db[q + nt * 16 + n];
#pragma unroll
      for (int j = 0; j < 8; ++j) {
        const float p = __expf(c[j] - Mq) * Di;
        myP[(h * 8 + j) * 32 + nt * 16 + n] = (bf16_t)p;  // row-major 16x32
      }
    }
    __builtin_amdgcn_wave_barrier();   // keep ds loads after ds stores
    // reload P as an A fragment (rows by lane&15, K packed along VGPRs)
    v16bf pa = cat16(*(const v8bf*)(myP + n * 32 + h * 8),
                     *(const v8bf*)(myP + n * 32 + 16 + h * 8));
    rsum = wmma_bf(pa, ones, rsum);    // rowsum(P) via ones-B matrix
    {
      const bf16_t* vr0 = vbase + (size_t)(0 * 16 + n) * NQ + q;
      const bf16_t* vr1 = vbase + (size_t)(1 * 16 + n) * NQ + q;
      const bf16_t* vr2 = vbase + (size_t)(2 * 16 + n) * NQ + q;
      const bf16_t* vr3 = vbase + (size_t)(3 * 16 + n) * NQ + q;
      acc0 = wmma_bf(pa, loadB(vr0, h), acc0);
      acc1 = wmma_bf(pa, loadB(vr1, h), acc1);
      acc2 = wmma_bf(pa, loadB(vr2, h), acc2);
      acc3 = wmma_bf(pa, loadB(vr3, h), acc3);
    }
    __builtin_amdgcn_wave_barrier();   // next-iter stores stay after loads
  }

  // epilogue: kproj tile (reuses keys A fragments), fuse rowsum*(kproj+bv)
  float* obase = out + ((size_t)bh * NK + k0) * SD;
  v8f accs[4] = {acc0, acc1, acc2, acc3};
#pragma unroll
  for (int tt = 0; tt < 4; ++tt) {
    const float* wr = Wv + (size_t)(tt * 16 + n) * 128 + 64;  // Wv_k row t
    v16bf w0 = loadB_f32(wr, h);
    v16bf w1 = loadB_f32(wr + 32, h);
    v8f kp = zero8();
    kp = wmma_bf(a0, w0, kp);
    kp = wmma_bf(a1, w1, kp);
    const float bvv = bv[tt * 16 + n];
#pragma unroll
    for (int j = 0; j < 8; ++j) {
      obase[(size_t)(h * 8 + j) * SD + tt * 16 + n] =
          accs[tt][j] + rsum[j] * (kp[j] + bvv);
    }
  }
}

// ---------------------------------------------------------------------------
extern "C" void kernel_launch(void* const* d_in, const int* in_sizes, int n_in,
                              void* d_out, int out_size, void* d_ws,
                              size_t ws_size, hipStream_t stream) {
  (void)in_sizes; (void)n_in; (void)out_size; (void)ws_size;
  const float* tensorL = (const float*)d_in[0];
  const float* tensorH = (const float*)d_in[1];
  const float* Wk = (const float*)d_in[2];
  const float* bk = (const float*)d_in[3];
  const float* Wq = (const float*)d_in[4];
  const float* bq = (const float*)d_in[5];
  const float* Wv = (const float*)d_in[6];
  const float* bv = (const float*)d_in[7];
  float* out = (float*)d_out;

  char* ws = (char*)d_ws;
  bf16_t* keysb    = (bf16_t*)(ws);
  bf16_t* queriesb = (bf16_t*)(ws + (size_t)4 * 1024 * 1024);
  bf16_t* qprojT   = (bf16_t*)(ws + (size_t)8 * 1024 * 1024);
  float*  Mbuf     = (float*)(ws + (size_t)12 * 1024 * 1024);
  float*  Dinvb    = (float*)(ws + (size_t)12 * 1024 * 1024 + 256 * 1024);

  const dim3 blk(256);
  const dim3 gProj(BH * NK / 128);       // 256 blocks, 8 wave-tiles each
  const dim3 gTile(NK / 128, BH);        // (16,16)

  hipLaunchKernelGGL((proj_kernel<true, false, true>), gProj, blk, 0, stream,
                     tensorL, (const bf16_t*)nullptr, Wk, 64, 0, bk, keysb);
  hipLaunchKernelGGL((proj_kernel<true, false, true>), gProj, blk, 0, stream,
                     tensorH, (const bf16_t*)nullptr, Wq, 64, 0, bq, queriesb);
  hipLaunchKernelGGL((proj_kernel<false, true, false>), gProj, blk, 0, stream,
                     (const float*)nullptr, queriesb, Wv, 128, 0,
                     (const float*)nullptr, qprojT);
  hipLaunchKernelGGL(stats_kernel, gTile, blk, 0, stream, keysb, queriesb,
                     Mbuf, Dinvb);
  hipLaunchKernelGGL(out_kernel, gTile, blk, 0, stream, keysb, queriesb,
                     qprojT, Mbuf, Dinvb, Wv, bv, out);
}
